// LaplaceEncoder_83021717831744
// MI455X (gfx1250) — matrix operand order repaired and checked by reading, not compiled, not run on hardware
//
#include <hip/hip_runtime.h>
#include <hip/hip_bf16.h>

typedef __attribute__((ext_vector_type(16))) _Float16 v16h;
typedef __attribute__((ext_vector_type(8)))  float    v8f;
typedef __attribute__((ext_vector_type(4)))  float    v4f;
typedef __attribute__((ext_vector_type(4)))  int      v4i;

#define B_  64
#define T_  1024
#define C_  64
#define H_  32
#define K_  10

union Frag { v16h h; v4i d[2]; };

// Branch-free unrolled sorted insertion into 10-entry ascending list (regs only).
#define INSERT10(dv, iv) do {                                        \
    float cd_ = (dv); int ci_ = (iv);                                \
    _Pragma("unroll")                                                \
    for (int s_ = 0; s_ < K_; ++s_) {                                \
      bool sw_ = cd_ < bestD[s_];                                    \
      float td_ = bestD[s_]; int ti_ = bestI[s_];                    \
      bestD[s_] = sw_ ? cd_ : td_;  bestI[s_] = sw_ ? ci_ : ti_;     \
      cd_ = sw_ ? td_ : cd_;        ci_ = sw_ ? ti_ : ti_ == ti_ ? ci_ : ci_; \
      ci_ = sw_ ? ti_ : ci_;                                         \
    } } while (0)

// (note: the stray ternary above would be wrong; use clean macro below instead)
#undef INSERT10
#define INSERT10(dv, iv) do {                                        \
    float cd_ = (dv); int ci_ = (iv);                                \
    _Pragma("unroll")                                                \
    for (int s_ = 0; s_ < K_; ++s_) {                                \
      bool sw_ = cd_ < bestD[s_];                                    \
      float td_ = bestD[s_]; int ti_ = bestI[s_];                    \
      bestD[s_] = sw_ ? cd_ : td_;  bestI[s_] = sw_ ? ci_ : ti_;     \
      cd_ = sw_ ? td_ : cd_;        ci_ = sw_ ? ti_ : ci_;           \
    } } while (0)

// ---------------- Kernel 1: h = x@W1 + b1 ; h16 = (f16)h ; sqn = ||h||^2 ----
__global__ __launch_bounds__(256) void k_embed(
    const float* __restrict__ x, const float* __restrict__ W1,
    const float* __restrict__ b1,
    float* __restrict__ h, _Float16* __restrict__ h16, float* __restrict__ sqn)
{
  __shared__ float sW1[C_ * H_];
  __shared__ float sb1[H_];
  int tid = threadIdx.x;
  for (int i = tid; i < C_ * H_; i += 256) sW1[i] = W1[i];
  if (tid < H_) sb1[tid] = b1[tid];
  __syncthreads();

  int row = blockIdx.x * 256 + tid;               // 0 .. B*T-1
  const float* xr = x + (size_t)row * C_;

  float acc[H_];
#pragma unroll
  for (int o = 0; o < H_; ++o) acc[o] = sb1[o];
  for (int c = 0; c < C_; ++c) {
    float xv = xr[c];
#pragma unroll
    for (int o = 0; o < H_; ++o) acc[o] = fmaf(xv, sW1[c * H_ + o], acc[o]);
  }

  float sq = 0.f;
#pragma unroll
  for (int o = 0; o < H_; ++o) sq = fmaf(acc[o], acc[o], sq);

  float* hr = h + (size_t)row * H_;
#pragma unroll
  for (int o = 0; o < H_; o += 4)
    *(float4*)(hr + o) = make_float4(acc[o], acc[o+1], acc[o+2], acc[o+3]);

  unsigned* pr = (unsigned*)(h16 + (size_t)row * H_);
#pragma unroll
  for (int p = 0; p < H_ / 2; ++p) {
    union { _Float16 f[2]; unsigned u; } cv;
    cv.f[0] = (_Float16)acc[2 * p];
    cv.f[1] = (_Float16)acc[2 * p + 1];
    pr[p] = cv.u;
  }
  sqn[row] = sq;
}

// ---------------- Kernel 2: WMMA Gram tiles -> top-10 -> gaussian smooth -----
__global__ __launch_bounds__(128) void k_knn_fused(
    const float* __restrict__ h, const _Float16* __restrict__ h16,
    const float* __restrict__ sqn,
    const float* __restrict__ W2, const float* __restrict__ b2,
    float* __restrict__ out)
{
  __shared__ float sW2[H_ * H_];
  __shared__ float sb2[H_];
  __shared__ float sdist[4][16][20];     // per-wave 16x16 d2 tile, row pad 20 -> b128-aligned halves
  __shared__ float smD[4][32][K_];
  __shared__ int   smI[4][32][K_];

  int tid  = threadIdx.x;
  int wave = tid >> 5;
  int lane = tid & 31;
  int blk  = blockIdx.x;                 // 0..1023
  int b    = blk >> 4;                   // batch
  int rb   = blk & 15;                   // 64-row block within batch
  int rowBase = rb * 64 + wave * 16;     // this wave's 16 rows (t index)

  const float*    hB   = h   + (size_t)b * T_ * H_;
  const _Float16* h16B = h16 + (size_t)b * T_ * H_;
  const float*    sqB  = sqn + (size_t)b * T_;

  for (int i = tid; i < H_ * H_; i += 128) sW2[i] = W2[i];
  if (tid < H_) sb2[tid] = b2[tid];
  __syncthreads();

  int lh = lane >> 4;                    // lane-half (0/1)
  int ll = lane & 15;

  // A fragment: 16x32 f16 (lanes 0-15: K=[0..7],[16..23]; lanes 16-31: K=[8..15],[24..31])
  Frag a;
  {
    const _Float16* ap = h16B + (size_t)(rowBase + ll) * H_;
    int ks = lh * 8;
    a.d[0] = *(const v4i*)(ap + ks);
    a.d[1] = *(const v4i*)(ap + ks + 16);
  }

  int mbase = rowBase + lh * 8;          // global row for C slot r: mbase + r
  float sqi[8];
#pragma unroll
  for (int r = 0; r < 8; ++r) sqi[r] = sqB[mbase + r];

  float bestD[K_]; int bestI[K_];
#pragma unroll
  for (int s = 0; s < K_; ++s) { bestD[s] = 3.0e38f; bestI[s] = 0; }

  int selRow  = lane >> 1;               // 2 lanes per row
  int selCol0 = (lane & 1) * 8;

  for (int jt = 0; jt < 64; ++jt) {
    // B fragment: 32x16 f16, lane holds column N=ll, K=[8h..8h+7],[16+8h..16+8h+7]
    Frag bf;
    const _Float16* bp = h16B + (size_t)(jt * 16 + ll) * H_ + lh * 8;
    bf.d[0] = *(const v4i*)(bp);
    bf.d[1] = *(const v4i*)(bp + 16);
    if (jt + 1 < 64)
      __builtin_prefetch(h16B + (size_t)((jt + 1) * 16 + ll) * H_, 0, 1);

    v8f acc = {};
    acc = __builtin_amdgcn_wmma_f32_16x16x32_f16(
        /*neg_a=*/false, a.h, /*neg_b=*/false, bf.h,
        /*c_mod=*/(short)0, acc, /*reuse_a=*/false, /*reuse_b=*/false);

    float sqj  = sqB[jt * 16 + ll];
    int   jcol = jt * 16 + ll;

    __builtin_amdgcn_wave_barrier();
#pragma unroll
    for (int r = 0; r < 8; ++r) {
      float d2 = fmaxf(sqi[r] + sqj - 2.0f * acc[r], 0.0f);
      if (mbase + r == jcol) d2 = 1.0e18f;          // never self
      sdist[wave][lh * 8 + r][ll] = d2;
    }
    __builtin_amdgcn_wave_barrier();

    // read my 8 candidates as two b128 LDS loads
    v4f c0 = *(const v4f*)&sdist[wave][selRow][selCol0];
    v4f c1 = *(const v4f*)&sdist[wave][selRow][selCol0 + 4];
    float cand[8] = {c0.x, c0.y, c0.z, c0.w, c1.x, c1.y, c1.z, c1.w};

    // tile-level early-out: only enter insert block if some candidate improves
    float m01 = fminf(cand[0], cand[1]);
    float m23 = fminf(cand[2], cand[3]);
    float m45 = fminf(cand[4], cand[5]);
    float m67 = fminf(cand[6], cand[7]);
    float m8  = fminf(fminf(m01, m23), fminf(m45, m67));
    if (m8 < bestD[K_ - 1]) {
#pragma unroll
      for (int n = 0; n < 8; ++n) {
        float d = cand[n];
        int   j = jt * 16 + selCol0 + n;
        if (d < bestD[K_ - 1]) { INSERT10(d, j); }
      }
    }
    __builtin_amdgcn_wave_barrier();
  }

  // merge the two half-lists of each row
#pragma unroll
  for (int s = 0; s < K_; ++s) { smD[wave][lane][s] = bestD[s]; smI[wave][lane][s] = bestI[s]; }
  __builtin_amdgcn_wave_barrier();

  if ((lane & 1) == 0) {
#pragma unroll
    for (int s = 0; s < K_; ++s) {
      float d = smD[wave][lane + 1][s];
      int   j = smI[wave][lane + 1][s];
      if (d < bestD[K_ - 1]) { INSERT10(d, j); }
    }

    int i = rowBase + selRow;
    const float* hip_ = hB + (size_t)i * H_;
    float hi[H_], sm[H_];
#pragma unroll
    for (int c2 = 0; c2 < H_; ++c2) { hi[c2] = hip_[c2]; sm[c2] = 0.f; }

    float wsum = 0.f;
    const float wfac = -1.0f / (2.0f + 1e-8f);      // -1/(2*sigma^2+eps)
#pragma unroll
    for (int k2 = 0; k2 < K_; ++k2) {
      const float* np = hB + (size_t)bestI[k2] * H_;
      float nb[H_];
#pragma unroll
      for (int c2 = 0; c2 < H_; ++c2) nb[c2] = np[c2];
      float d2 = 0.f;
#pragma unroll
      for (int c2 = 0; c2 < H_; ++c2) { float df = nb[c2] - hi[c2]; d2 = fmaf(df, df, d2); }
      float w = __expf(d2 * wfac);                  // exact f32 recompute
      wsum += w;
#pragma unroll
      for (int c2 = 0; c2 < H_; ++c2) sm[c2] = fmaf(w, nb[c2], sm[c2]);
    }
    float inv = 1.0f / (wsum + 1e-8f);

    float tv[H_];
#pragma unroll
    for (int c2 = 0; c2 < H_; ++c2) tv[c2] = tanhf(hi[c2] - sm[c2] * inv);

    float* op = out + (size_t)(b * T_ + i) * H_;
#pragma unroll
    for (int o = 0; o < H_; ++o) {
      float accv = sb2[o];
#pragma unroll
      for (int c2 = 0; c2 < H_; ++c2) accv = fmaf(tv[c2], sW2[c2 * H_ + o], accv);
      op[o] = accv;
    }
  }
}

extern "C" void kernel_launch(void* const* d_in, const int* in_sizes, int n_in,
                              void* d_out, int out_size, void* d_ws, size_t ws_size,
                              hipStream_t stream) {
  (void)in_sizes; (void)n_in; (void)out_size; (void)ws_size;
  const float* x  = (const float*)d_in[0];
  const float* W1 = (const float*)d_in[1];
  const float* b1 = (const float*)d_in[2];
  const float* W2 = (const float*)d_in[3];
  const float* b2 = (const float*)d_in[4];
  float* out = (float*)d_out;

  char* ws = (char*)d_ws;
  float*    h   = (float*)ws;                                    // 8 MB
  _Float16* h16 = (_Float16*)(ws + (size_t)B_ * T_ * H_ * 4);    // 4 MB
  float*    sqn = (float*)(ws + (size_t)B_ * T_ * H_ * 6);       // 256 KB

  k_embed<<<(B_ * T_) / 256, 256, 0, stream>>>(x, W1, b1, h, h16, sqn);
  k_knn_fused<<<B_ * 16, 128, 0, stream>>>(h, h16, sqn, W2, b2, out);
}
